// whole_network_77919296684408
// MI455X (gfx1250) — matrix operand order repaired and checked by reading, not compiled, not run on hardware
//
#include <hip/hip_runtime.h>

#define BB 4096
#define RR 130
#define TT 200
#define QQ 64
#define RELP 160   // padded column stride for f16 relation matrix (5 * K32 steps)
#define EPSV 1e-5f

typedef __attribute__((ext_vector_type(16))) _Float16 v16h;
typedef __attribute__((ext_vector_type(8)))  _Float16 v8h;
typedef __attribute__((ext_vector_type(2)))  _Float16 v2h;
typedef __attribute__((ext_vector_type(8)))  float    v8f;
typedef unsigned int u32;
typedef __attribute__((ext_vector_type(4))) u32 u32x4;
typedef __attribute__((ext_vector_type(8))) int i32x8;
typedef __attribute__((ext_vector_type(4))) int i32x4;

#if __has_builtin(__builtin_amdgcn_tensor_load_to_lds) && __has_builtin(__builtin_amdgcn_s_wait_tensorcnt)
#define HAVE_TDM 1
#else
#define HAVE_TDM 0
#endif

#if HAVE_TDM
// 1D TDM copy: n_half f16 elements from global to LDS (n_half < 65536).
// D# built per CDNA5 ISA 8.3/8.4: group0 = {count=1 | lds_addr | global_addr | type=2},
// group1 = {data_size=2B, tensor_dim0=n_half, tensor_dim1=1, tile_dim0=n_half, tile_dim1=1,
//           tensor_dim0_stride=n_half}. Groups 2/3 + 5D group zero (<=2D tensor).
static __device__ inline void tdm_load_1d(u32 lds_off, const void* gptr, u32 n_half) {
  unsigned long long ga = (unsigned long long)gptr;
  u32x4 g0;
  g0[0] = 1u;                                             // count=1 (valid user descriptor)
  g0[1] = lds_off;                                        // lds_addr
  g0[2] = (u32)ga;                                        // global_addr[31:0]
  g0[3] = (u32)((ga >> 32) & 0x01FFFFFFu) | (2u << 30);   // global_addr[56:32], type=2
  i32x8 g1;
  g1[0] = (int)(1u << 16);                                // data_size=1 -> 2 bytes
  g1[1] = (int)((n_half & 0xFFFFu) << 16);                // tensor_dim0[15:0]
  g1[2] = (int)(((n_half >> 16) & 0xFFFFu) | (1u << 16)); // tensor_dim0[31:16], tensor_dim1=1
  g1[3] = (int)((n_half & 0xFFFFu) << 16);                // tile_dim0 = n_half
  g1[4] = 1;                                              // tile_dim1 = 1
  g1[5] = (int)n_half;                                    // tensor_dim0_stride[31:0]
  g1[6] = 0;
  g1[7] = 0;
  i32x4 z4 = {0, 0, 0, 0};
  i32x8 z8 = {0, 0, 0, 0, 0, 0, 0, 0};
  __builtin_amdgcn_tensor_load_to_lds(g0, g1, z4, z4, z8, 0);
}
#endif

static __device__ inline v16h frag_zero() {
  v16h a;
#pragma unroll
  for (int e = 0; e < 16; ++e) a[e] = (_Float16)0.0f;
  return a;
}

// A fragment (16x32 f16) gathered from row-major f32 global src (NaN -> 0), row stride ld.
static __device__ inline v16h frag_a_f32(const float* __restrict__ src, int ld, int m0,
                                         int k0, int mMax, int kMax, int lane) {
  int m = m0 + (lane & 15);
  int half = lane >> 4;
  v16h a = frag_zero();
  if (m >= mMax) return a;
  const float* row = src + (size_t)m * ld;
#pragma unroll
  for (int p = 0; p < 8; ++p) {
    int k = k0 + ((p & 4) << 2) + (half << 3) + ((p & 3) << 1);
    float v0 = 0.f, v1 = 0.f;
    if (k < kMax)     { v0 = row[k];     if (!(v0 == v0)) v0 = 0.f; }
    if (k + 1 < kMax) { v1 = row[k + 1]; if (!(v1 == v1)) v1 = 0.f; }
    a[2 * p]     = (_Float16)v0;
    a[2 * p + 1] = (_Float16)v1;
  }
  return a;
}

// A fragment from row-major f16 LDS tile: per-lane K runs are contiguous -> 2x ds_load_b128.
static __device__ inline v16h frag_a_lds(const _Float16* sm, int ld, int m0, int k0,
                                         int mMax, int lane) {
  int m = m0 + (lane & 15);
  int base = k0 + ((lane >> 4) << 3);
  v16h a = frag_zero();
  if (m >= mMax) return a;
  const _Float16* row = sm + (size_t)m * ld + base;
  v8h lo = *(const v8h*)(row);
  v8h hi = *(const v8h*)(row + 16);
#pragma unroll
  for (int e = 0; e < 8; ++e) { a[e] = lo[e]; a[8 + e] = hi[e]; }
  return a;
}

// A fragment from global row-major f16 with exactly 16 columns (h16): one global_load_b128,
// upper K half (16..31) is zero padding.
static __device__ inline v16h frag_a_h16(const _Float16* __restrict__ src, int m0,
                                         int mMax, int lane) {
  int m = m0 + (lane & 15);
  int base = (lane >> 4) << 3;
  v16h a = frag_zero();
  if (m >= mMax) return a;
  v8h lo = *(const v8h*)(src + (size_t)m * 16 + base);
#pragma unroll
  for (int e = 0; e < 8; ++e) a[e] = lo[e];
  return a;
}

// Pair-major LDS B tile: element (k, n) stored at ((k/2)*2*ld + 2n + (k&1)).
static __device__ inline int sB(int k, int n, int ld) {
  return (k >> 1) * (2 * ld) + 2 * n + (k & 1);
}

// B fragment (32x16 f16) from pair-major LDS tile: 8x ds_load_b32.
static __device__ inline v16h frag_b_pm(const _Float16* sm, int ld, int k0, int n0, int lane) {
  int n = n0 + (lane & 15);
  int kb = k0 + ((lane >> 4) << 3);  // even
  v16h b;
#pragma unroll
  for (int p = 0; p < 8; ++p) {
    v2h pr = *(const v2h*)(sm + (size_t)((kb >> 1) + (p & 3) + ((p & 4) << 1)) * (2 * ld) + 2 * n);
    b[2 * p]     = pr[0];
    b[2 * p + 1] = pr[1];
  }
  return b;
}

// B fragment for faithful reshape Km (64x130): Km[k][n] = qk_flat[k*130 + n], from LDS.
static __device__ inline v16h frag_b_km_lds(const _Float16* sm, int k0, int n0, int lane) {
  int n = n0 + (lane & 15);
  int kb = k0 + ((lane >> 4) << 3);
  v16h b = frag_zero();
  if (n >= RR) return b;
#pragma unroll
  for (int p = 0; p < 8; ++p) {
    int base = (kb + 2 * (p & 3) + ((p & 4) << 2)) * RR + n;
    b[2 * p]     = sm[base];
    b[2 * p + 1] = sm[base + RR];
  }
  return b;
}

// ---------------------------------------------------------------- qk = x @ wq^T + bq
__global__ __launch_bounds__(128) void k_qk(const float* __restrict__ bold,
                                            const float* __restrict__ wq,
                                            const float* __restrict__ bq,
                                            _Float16* __restrict__ qk16) {
  __shared__ __align__(16) _Float16 Bsm[224 * QQ];  // wq^T, pair-major f16, K padded
  int mt = blockIdx.x;                // 0..8 M tiles over R
  int b  = blockIdx.y;
  int tid = threadIdx.x, lane = tid & 31, wv = tid >> 5;  // 4 waves, one N tile each
  for (int idx = tid; idx < 224 * QQ; idx += 128) {
    int k = idx >> 6, n = idx & 63;
    Bsm[sB(k, n, QQ)] = (k < TT) ? (_Float16)wq[n * TT + k] : (_Float16)0.0f;
  }
  __syncthreads();
  const float* xb = bold + (size_t)b * RR * TT;
  __builtin_prefetch(xb + (size_t)(mt * 16) * TT, 0, 1);
  v8f acc = {};
#pragma unroll
  for (int kt = 0; kt < 7; ++kt) {
    v16h a  = frag_a_f32(xb, TT, mt * 16, kt * 32, RR, TT, lane);
    v16h bf = frag_b_pm(Bsm, QQ, kt * 32, wv * 16, lane);
    acc = __builtin_amdgcn_wmma_f32_16x16x32_f16(false, a, false, bf, (short)0, acc, false, false);
  }
  int half = lane >> 4, col = wv * 16 + (lane & 15);
#pragma unroll
  for (int v = 0; v < 8; ++v) {
    int row = mt * 16 + v + 8 * half;
    if (row < RR)
      qk16[(size_t)b * RR * QQ + (size_t)row * QQ + col] = (_Float16)(acc[v] + bq[col]);
  }
}

// -------- rel_pre = Qm @ Km, one block per batch; qk tile staged to LDS via TDM;
// -------- fused BN partial sums via LDS float atomics.
__global__ __launch_bounds__(288) void k_rel(const _Float16* __restrict__ qk16,
                                             float* __restrict__ relpre,
                                             float* __restrict__ sums,
                                             float* __restrict__ sumsq) {
  __shared__ __align__(16) _Float16 qsm[RR * QQ];   // 8320 halves, row-major (130x64)
  __shared__ float ssum[144], ssq[144];
  int b = blockIdx.x;
  int tid = threadIdx.x, lane = tid & 31, wv = tid >> 5;  // 9 waves, one M tile each
  const _Float16* qkb = qk16 + (size_t)b * RR * QQ;
#if HAVE_TDM
  if (wv == 0) tdm_load_1d((u32)(unsigned long long)&qsm[0], qkb, RR * QQ);
#else
  {
    const uint4* gs = (const uint4*)qkb;
    uint4* ds = (uint4*)&qsm[0];
    for (int i = tid; i < RR * QQ / 8; i += 288) ds[i] = gs[i];
  }
#endif
  for (int idx = tid; idx < 144; idx += 288) { ssum[idx] = 0.f; ssq[idx] = 0.f; }
#if HAVE_TDM
  if (wv == 0) __builtin_amdgcn_s_wait_tensorcnt((short)0);
#endif
  __syncthreads();

  int mt = wv;
  float* relb = relpre + (size_t)b * RR * RR;
  for (int nt = 0; nt < 9; ++nt) {
    v8f acc = {};
#pragma unroll
    for (int kt = 0; kt < 2; ++kt) {
      v16h a  = frag_a_lds(qsm, QQ, mt * 16, kt * 32, RR, lane);
      v16h bf = frag_b_km_lds(qsm, kt * 32, nt * 16, lane);
      acc = __builtin_amdgcn_wmma_f32_16x16x32_f16(false, a, false, bf, (short)0, acc, false, false);
    }
    int half = lane >> 4, col = nt * 16 + (lane & 15);
#pragma unroll
    for (int v = 0; v < 8; ++v) {
      int row = mt * 16 + v + 8 * half;
      if (row < RR && col < RR) {
        float val = acc[v];
        relb[(size_t)row * RR + col] = val;
        atomicAdd(&ssum[row], val);
        atomicAdd(&ssq[row], val * val);
      }
    }
  }
  __syncthreads();
  if (tid < RR) { atomicAdd(&sums[tid], ssum[tid]); atomicAdd(&sumsq[tid], ssq[tid]); }
}

// ------------------------------------------------------- BN stats -> per-channel affine
__global__ void k_bnfin(const float* __restrict__ sums, const float* __restrict__ sumsq,
                        float cnt, const float* __restrict__ g, const float* __restrict__ beta,
                        float* __restrict__ scale, float* __restrict__ shift) {
  int r = threadIdx.x;
  if (r < RR) {
    float mean = sums[r] / cnt;
    float var  = sumsq[r] / cnt - mean * mean;
    float sc   = g[r] * rsqrtf(var + EPSV);
    scale[r] = sc;
    shift[r] = beta[r] - mean * sc;
  }
}

// ------------------------------- affine + softmax(axis2) + relu(x - thr), write f16 padded
__global__ __launch_bounds__(256) void k_softmax(const float* __restrict__ relpre,
                                                 const float* __restrict__ scale,
                                                 const float* __restrict__ shift,
                                                 const float* __restrict__ thr,
                                                 _Float16* __restrict__ rel16) {
  int wid = (blockIdx.x << 3) + (threadIdx.x >> 5);  // one wave per (b, r) row
  if (wid >= BB * RR) return;
  int lane = threadIdx.x & 31;
  int r = wid % RR;
  const float* row = relpre + (size_t)wid * RR;
  float sc = scale[r], sh = shift[r];
  float x[5];
#pragma unroll
  for (int i = 0; i < 5; ++i) {
    int s = lane + 32 * i;
    x[i] = (s < RR) ? (row[s] * sc + sh) : -3.0e38f;
  }
  float m = x[0];
#pragma unroll
  for (int i = 1; i < 5; ++i) m = fmaxf(m, x[i]);
#pragma unroll
  for (int off = 16; off > 0; off >>= 1) m = fmaxf(m, __shfl_xor(m, off, 32));
  float e[5], sum = 0.f;
#pragma unroll
  for (int i = 0; i < 5; ++i) {
    int s = lane + 32 * i;
    e[i] = (s < RR) ? __expf(x[i] - m) : 0.f;
    sum += e[i];
  }
#pragma unroll
  for (int off = 16; off > 0; off >>= 1) sum += __shfl_xor(sum, off, 32);
  float inv = 1.0f / sum;
  float t = thr[0];
  _Float16* orow = rel16 + (size_t)wid * RELP;
#pragma unroll
  for (int i = 0; i < 5; ++i) {
    int s = lane + 32 * i;
    float o = fmaxf(e[i] * inv - t, 0.f);
    orow[s] = (s < RR) ? (_Float16)o : (_Float16)0.0f;
  }
}

// ---- GCN layer. Phase A: g = f @ w^T into LDS while TDM streams the relation tile in.
// ---- Phase B: y = rel @ g + bias from LDS; fused per-channel BN partial sums.
__global__ __launch_bounds__(288) void k_gcn(const float* __restrict__ f32src,
                                             const _Float16* __restrict__ f16src,
                                             int use_f32, int fiK, int fi_true, int fo,
                                             const float* __restrict__ w,
                                             const float* __restrict__ bias,
                                             const _Float16* __restrict__ rel16,
                                             float* __restrict__ ybuf,
                                             float* __restrict__ sums,
                                             float* __restrict__ sumsq) {
  __shared__ __align__(16) _Float16 relsm[RR * RELP];   // 130x160 f16 = 41.6 KB
  __shared__ __align__(16) _Float16 wsm[224 * 16];      // w^T, pair-major, zero padded
  __shared__ __align__(16) _Float16 gsm[RELP * 16];     // g, pair-major, rows padded
  __shared__ float ssum[144], ssq[144];
  int b = blockIdx.x;
  int tid = threadIdx.x, lane = tid & 31, wv = tid >> 5;  // 9 waves, one M tile each

#if HAVE_TDM
  if (wv == 0)
    tdm_load_1d((u32)(unsigned long long)&relsm[0], rel16 + (size_t)b * RR * RELP, RR * RELP);
#else
  {
    const uint4* gs = (const uint4*)(rel16 + (size_t)b * RR * RELP);
    uint4* ds = (uint4*)&relsm[0];
    for (int i = tid; i < RR * RELP / 8; i += 288) ds[i] = gs[i];
  }
#endif
  for (int idx = tid; idx < fiK * 16; idx += 288) {
    int k = idx >> 4, n = idx & 15;
    wsm[sB(k, n, 16)] = (k < fi_true && n < fo) ? (_Float16)w[n * fi_true + k] : (_Float16)0.0f;
  }
  for (int idx = tid; idx < (RELP - 144) * 16; idx += 288) {
    int k = 144 + (idx >> 4), n = idx & 15;
    gsm[sB(k, n, 16)] = (_Float16)0.0f;
  }
  for (int idx = tid; idx < 144; idx += 288) { ssum[idx] = 0.f; ssq[idx] = 0.f; }
  __syncthreads();

  {  // phase A: g = f @ w^T  (overlaps with the TDM transfer of relsm)
    v8f acc = {};
    if (use_f32) {
      for (int kt = 0; kt < fiK / 32; ++kt) {
        v16h a  = frag_a_f32(f32src + (size_t)b * RR * TT, TT, wv * 16, kt * 32, RR, TT, lane);
        v16h bf = frag_b_pm(wsm, 16, kt * 32, 0, lane);
        acc = __builtin_amdgcn_wmma_f32_16x16x32_f16(false, a, false, bf, (short)0, acc, false, false);
      }
    } else {
      v16h a  = frag_a_h16(f16src + (size_t)b * RR * 16, wv * 16, RR, lane);
      v16h bf = frag_b_pm(wsm, 16, 0, 0, lane);
      acc = __builtin_amdgcn_wmma_f32_16x16x32_f16(false, a, false, bf, (short)0, acc, false, false);
    }
    int half = lane >> 4, col = lane & 15;
#pragma unroll
    for (int v = 0; v < 8; ++v) {
      int row = wv * 16 + v + 8 * half;
      gsm[sB(row, col, 16)] = (_Float16)acc[v];
    }
  }
#if HAVE_TDM
  if (wv == 0) __builtin_amdgcn_s_wait_tensorcnt((short)0);
#endif
  __syncthreads();

  {  // phase B: y = rel @ g  (K = 160 zero padded), all operands in LDS
    v8f acc = {};
#pragma unroll
    for (int kt = 0; kt < 5; ++kt) {
      v16h a  = frag_a_lds(relsm, RELP, wv * 16, kt * 32, RR, lane);
      v16h bf = frag_b_pm(gsm, 16, kt * 32, 0, lane);
      acc = __builtin_amdgcn_wmma_f32_16x16x32_f16(false, a, false, bf, (short)0, acc, false, false);
    }
    int half = lane >> 4, col = lane & 15;
#pragma unroll
    for (int v = 0; v < 8; ++v) {
      int row = wv * 16 + v + 8 * half;
      if (row < RR) {
        float val = acc[v] + ((col < fo) ? bias[col] : 0.f);
        ybuf[((size_t)b * RR + row) * 16 + col] = val;
        if (col < fo) { atomicAdd(&ssum[row], val); atomicAdd(&ssq[row], val * val); }
      }
    }
  }
  __syncthreads();
  if (tid < RR) { atomicAdd(&sums[tid], ssum[tid]); atomicAdd(&sumsq[tid], ssq[tid]); }
}

// --------------------------------- bn apply + relu + optional residual; f32 + f16 copies
__global__ __launch_bounds__(256) void k_apply(const float* __restrict__ ybuf,
                                               const float* __restrict__ scale,
                                               const float* __restrict__ shift,
                                               int fo, int residual,
                                               const float* __restrict__ hprev,
                                               float* __restrict__ hout,
                                               _Float16* __restrict__ h16) {
  size_t i = (size_t)blockIdx.x * 256 + threadIdx.x;
  if (i >= (size_t)BB * RR * 16) return;
  int c = (int)(i & 15);
  int r = (int)((i >> 4) % RR);
  float v = 0.f;
  if (c < fo) {
    v = fmaxf(ybuf[i] * scale[r] + shift[r], 0.f);
    if (residual) v += hprev[i];
  }
  hout[i] = v;
  h16[i] = (_Float16)v;
}

// --------------------------------------------------------------- out = relu(h_flat @ Wm^T + bm)
__global__ __launch_bounds__(256) void k_mlp(const float* __restrict__ h,
                                             const float* __restrict__ wm,
                                             const float* __restrict__ bm,
                                             float* __restrict__ out) {
  int b = (blockIdx.x << 3) + (threadIdx.x >> 5);  // one wave per batch
  if (b >= BB) return;
  int lane = threadIdx.x & 31;
  float a0 = 0.f, a1 = 0.f;
  const float* hb = h + (size_t)b * RR * 16;
  for (int r = lane; r < RR; r += 32) {
    float h0 = hb[r * 16 + 0], h1 = hb[r * 16 + 1];
    a0 += h0 * wm[r * 2] + h1 * wm[r * 2 + 1];
    a1 += h0 * wm[2 * RR + r * 2] + h1 * wm[2 * RR + r * 2 + 1];
  }
#pragma unroll
  for (int off = 16; off > 0; off >>= 1) {
    a0 += __shfl_xor(a0, off, 32);
    a1 += __shfl_xor(a1, off, 32);
  }
  if (lane == 0) {
    out[b * 2 + 0] = fmaxf(a0 + bm[0], 0.f);
    out[b * 2 + 1] = fmaxf(a1 + bm[1], 0.f);
  }
}

extern "C" void kernel_launch(void* const* d_in, const int* in_sizes, int n_in,
                              void* d_out, int out_size, void* d_ws, size_t ws_size,
                              hipStream_t stream) {
  (void)in_sizes; (void)n_in; (void)out_size; (void)ws_size;
  const float* bold = (const float*)d_in[0];
  const float* thr  = (const float*)d_in[1];
  const float* wq   = (const float*)d_in[2];
  const float* bq   = (const float*)d_in[3];
  const float* g_t  = (const float*)d_in[4];
  const float* b_t  = (const float*)d_in[5];
  const float *wL[6], *bL[6], *gL[6], *betaL[6];
  for (int i = 0; i < 6; ++i) {
    wL[i]    = (const float*)d_in[6 + 4 * i];
    bL[i]    = (const float*)d_in[7 + 4 * i];
    gL[i]    = (const float*)d_in[8 + 4 * i];
    betaL[i] = (const float*)d_in[9 + 4 * i];
  }
  const float* wm = (const float*)d_in[30];
  const float* bm = (const float*)d_in[31];
  float* out = (float*)d_out;

  char* ws = (char*)d_ws;
  size_t off = 0;
  auto alloc = [&](size_t bytes) {
    char* p = ws + off;
    off = (off + bytes + 255) & ~(size_t)255;
    return p;
  };
  _Float16* qk16   = (_Float16*)alloc((size_t)BB * RR * QQ * 2);
  float*    relpre = (float*)   alloc((size_t)BB * RR * RR * 4);
  _Float16* rel16  = (_Float16*)alloc((size_t)BB * RR * RELP * 2);
  float*    ybuf   = (float*)   alloc((size_t)BB * RR * 16 * 4);
  float*    hA     = (float*)   alloc((size_t)BB * RR * 16 * 4);
  float*    hB     = (float*)   alloc((size_t)BB * RR * 16 * 4);
  _Float16* h16    = (_Float16*)alloc((size_t)BB * RR * 16 * 2);
  float*    stats  = (float*)   alloc(4 * RR * sizeof(float));
  float *sums = stats, *sumsq = stats + RR, *scale = stats + 2 * RR, *shift = stats + 3 * RR;

  // 1) qk projection
  k_qk<<<dim3(9, BB), 128, 0, stream>>>(bold, wq, bq, qk16);

  // 2) relation matrix + fused BN stats (one block per batch, TDM-staged qk tile)
  (void)hipMemsetAsync(sums, 0, 2 * RR * sizeof(float), stream);
  k_rel<<<BB, 288, 0, stream>>>(qk16, relpre, sums, sumsq);
  k_bnfin<<<1, 160, 0, stream>>>(sums, sumsq, (float)((size_t)BB * RR), g_t, b_t, scale, shift);

  // 3) affine + softmax + threshold -> f16 relation matrix
  k_softmax<<<(BB * RR) / 8, 256, 0, stream>>>(relpre, scale, shift, thr, rel16);

  // 4) six GCN layers
  const int fiT[6]  = {TT, 16, 16, 4, 4, 2};
  const int foT[6]  = {16, 16, 4, 4, 2, 2};
  const int fiK[6]  = {224, 32, 32, 32, 32, 32};
  const int resi[6] = {0, 1, 0, 1, 0, 1};
  float* hcur = hA;
  for (int L = 0; L < 6; ++L) {
    (void)hipMemsetAsync(sums, 0, 2 * RR * sizeof(float), stream);
    k_gcn<<<BB, 288, 0, stream>>>((L == 0) ? bold : nullptr,
                                  (L == 0) ? nullptr : (const _Float16*)h16,
                                  (L == 0) ? 1 : 0,
                                  fiK[L], fiT[L], foT[L],
                                  wL[L], bL[L], rel16, ybuf, sums, sumsq);
    k_bnfin<<<1, 160, 0, stream>>>(sums, sumsq, (float)((size_t)BB * foT[L]),
                                   gL[L], betaL[L], scale, shift);
    float* hout = (L & 1) ? hB : hA;
    const float* hprev = hcur;
    k_apply<<<(int)(((size_t)BB * RR * 16 + 255) / 256), 256, 0, stream>>>(
        ybuf, scale, shift, foT[L], resi[L], hprev, hout, h16);
    hcur = hout;
  }

  // 5) MLP head
  k_mlp<<<BB / 8, 256, 0, stream>>>(hcur, wm, bm, out);
}
